// Attention_1958505087704
// MI455X (gfx1250) — matrix operand order repaired and checked by reading, not compiled, not run on hardware
//
#include <hip/hip_runtime.h>
#include <stddef.h>

typedef __bf16 bf16_t;
typedef __attribute__((ext_vector_type(16))) __bf16 v16bf;
typedef __attribute__((ext_vector_type(8)))  __bf16 v8bf;
typedef __attribute__((ext_vector_type(8)))  float  v8f;

#define BATCH 64
#define SEQL  2048
#define EDIM  1024
#define ADIM  512

// ---------------------------------------------------------------------------
// Kernel 1: We_w [E,A] fp32  ->  Wt [A,E] bf16 (transposed, B-operand friendly)
// ---------------------------------------------------------------------------
__global__ __launch_bounds__(512) void we_transpose_kernel(
    const float* __restrict__ We_w, bf16_t* __restrict__ Wt)
{
    int e = blockIdx.x;      // 0..1023
    int a = threadIdx.x;     // 0..511
    Wt[(size_t)a * EDIM + e] = (bf16_t)We_w[(size_t)e * ADIM + a];
}

// ---------------------------------------------------------------------------
// Kernel 2: dec_attn[b,a] = dh[b,:] @ Wd[:,a] + Wd_b[a] + We_b[a]
// ---------------------------------------------------------------------------
__global__ __launch_bounds__(512) void dec_proj_kernel(
    const float* __restrict__ dh, const float* __restrict__ Wd_w,
    const float* __restrict__ Wd_b, const float* __restrict__ We_b,
    float* __restrict__ decc)
{
    int b = blockIdx.x;
    int a = threadIdx.x;
    const float* h = dh + (size_t)b * EDIM;
    float acc = Wd_b[a] + We_b[a];
    for (int e = 0; e < EDIM; ++e)
        acc += h[e] * Wd_w[(size_t)e * ADIM + a];
    decc[(size_t)b * ADIM + a] = acc;
}

// ---------------------------------------------------------------------------
// Kernel 3: fused scores.  WG = (32 L-rows, one b).  8 waves; each wave owns a
// 64-col N strip (4 n-tiles) and computes BOTH 16-row M tiles, so every B
// fragment is loaded once per WG (halves Wt L2 traffic vs M-split waves).
//   GEMM (bf16 WMMA, fp32 acc) + relu(+dec)·Wf reduction fused in registers.
// LDS A tile: 32 rows x 512 K-chunk, row stride 520 bf16 (bank-conflict free).
// ---------------------------------------------------------------------------
#define A_STRIDE 520

__global__ __launch_bounds__(256) void attn_scores_kernel(
    const float* __restrict__ enc, const bf16_t* __restrict__ Wt,
    const float* __restrict__ decc, const float* __restrict__ Wf_w,
    const float* __restrict__ Wf_b, float* __restrict__ scores)
{
    __shared__ float  sPart[2][8][16];
    __shared__ bf16_t Atile[32 * A_STRIDE];

    const int b      = blockIdx.y;
    const int l0     = blockIdx.x * 32;
    const int tid    = threadIdx.x;
    const int lane   = tid & 31;
    const int w      = tid >> 5;      // 0..7: owns N strip [w*64, w*64+64)
    const int lanelo = lane & 15;
    const int hi     = lane >> 4;

    v8f acc[2][4];
#pragma unroll
    for (int mt = 0; mt < 2; ++mt)
#pragma unroll
        for (int nt = 0; nt < 4; ++nt)
            acc[mt][nt] = (v8f){0.f,0.f,0.f,0.f,0.f,0.f,0.f,0.f};

    const float* encb = enc + ((size_t)b * SEQL + l0) * EDIM;

    for (int kc = 0; kc < EDIM; kc += 512) {
        __syncthreads();
        // cooperative coalesced load 32x512 fp32 -> bf16 LDS
#pragma unroll
        for (int it = 0; it < 16; ++it) {
            int i   = tid + it * 256;
            int row = i >> 7;          // 128 float4 per row
            int c4  = i & 127;
            float4 v = *(const float4*)(encb + (size_t)row * EDIM + kc + c4 * 4);
            struct alignas(8) B4 { bf16_t v[4]; } p;
            p.v[0] = (bf16_t)v.x; p.v[1] = (bf16_t)v.y;
            p.v[2] = (bf16_t)v.z; p.v[3] = (bf16_t)v.w;
            *(B4*)&Atile[row * A_STRIDE + c4 * 4] = p;
        }
        __syncthreads();

        if (kc == 0) {
            // warm L2 with the second K-chunk while chunk 0 computes
            // (512 cachelines of 128B; 2 per thread) -> global_prefetch_b8
#pragma unroll
            for (int it = 0; it < 2; ++it) {
                int i   = tid + it * 256;
                int row = i >> 4;
                int cl  = i & 15;
                __builtin_prefetch(encb + (size_t)row * EDIM + 512 + cl * 32, 0, 3);
            }
        }

#pragma unroll 4
        for (int k0 = 0; k0 < 512; k0 += 32) {
            // A fragments for both M tiles: 16x32 bf16, ISA layout
            // (lanes 0-15: K kb..kb+7 & 16+kb..; lanes 16-31: +8)
            const bf16_t* ap0 = &Atile[lanelo * A_STRIDE + k0 + hi * 8];
            const bf16_t* ap1 = ap0 + 16 * A_STRIDE;
            v8bf a0lo = *(const v8bf*)ap0;
            v8bf a0hi = *(const v8bf*)(ap0 + 16);
            v8bf a1lo = *(const v8bf*)ap1;
            v8bf a1hi = *(const v8bf*)(ap1 + 16);
            v16bf af0 = __builtin_shufflevector(a0lo, a0hi,
                0,1,2,3,4,5,6,7,8,9,10,11,12,13,14,15);
            v16bf af1 = __builtin_shufflevector(a1lo, a1hi,
                0,1,2,3,4,5,6,7,8,9,10,11,12,13,14,15);
#pragma unroll
            for (int nt = 0; nt < 4; ++nt) {
                int ncol = w * 64 + nt * 16 + lanelo;
                // B fragment: 32x16 bf16, lane=col, K = kb..kb+7 & 16+kb..; lanes 16-31: +8
                const bf16_t* bp = Wt + (size_t)ncol * EDIM + kc + k0 + hi * 8;
                v8bf blo = *(const v8bf*)bp;
                v8bf bhi = *(const v8bf*)(bp + 16);
                v16bf bfrag = __builtin_shufflevector(blo, bhi,
                    0,1,2,3,4,5,6,7,8,9,10,11,12,13,14,15);
                acc[0][nt] = __builtin_amdgcn_wmma_f32_16x16x32_bf16(
                    false, af0, false, bfrag, (short)0, acc[0][nt], false, false);
                acc[1][nt] = __builtin_amdgcn_wmma_f32_16x16x32_bf16(
                    false, af1, false, bfrag, (short)0, acc[1][nt], false, false);
            }
        }
    }

    // epilogue: relu(enc_attn + dec_attn) . Wf, reduced over this wave's cols
    const float* db = decc + (size_t)b * ADIM;
    float sacc[2][8];
#pragma unroll
    for (int mt = 0; mt < 2; ++mt)
#pragma unroll
        for (int r = 0; r < 8; ++r) sacc[mt][r] = 0.f;
#pragma unroll
    for (int nt = 0; nt < 4; ++nt) {
        int ncol  = w * 64 + nt * 16 + lanelo;
        float dec = db[ncol];
        float wf  = Wf_w[ncol];
#pragma unroll
        for (int mt = 0; mt < 2; ++mt)
#pragma unroll
            for (int r = 0; r < 8; ++r) {
                float h = acc[mt][nt][r] + dec;  // C layout: M = r + 8*hi (+16*mt)
                h = h > 0.f ? h : 0.f;
                sacc[mt][r] += h * wf;
            }
    }
    // reduce across the 16 lanes of each half-wave (same M set, different N)
#pragma unroll
    for (int off = 8; off >= 1; off >>= 1)
#pragma unroll
        for (int mt = 0; mt < 2; ++mt)
#pragma unroll
            for (int r = 0; r < 8; ++r)
                sacc[mt][r] += __shfl_xor(sacc[mt][r], off, 32);
    if (lanelo == 0) {
#pragma unroll
        for (int mt = 0; mt < 2; ++mt)
#pragma unroll
            for (int r = 0; r < 8; ++r)
                sPart[mt][w][hi * 8 + r] = sacc[mt][r];
    }
    __syncthreads();
    if (tid < 32) {
        int i = tid & 15, t = tid >> 4;
        float s = Wf_b[0];
#pragma unroll
        for (int w2 = 0; w2 < 8; ++w2) s += sPart[t][w2][i];
        scores[(size_t)b * SEQL + l0 + tid] = s;
    }
}

// ---------------------------------------------------------------------------
// Kernel 4: softmax over L per batch row
// ---------------------------------------------------------------------------
__global__ __launch_bounds__(256) void softmax_kernel(
    const float* __restrict__ scores, float* __restrict__ alpha)
{
    __shared__ float red[8];
    __shared__ float bcast[2];
    int b = blockIdx.x, tid = threadIdx.x;
    const float* s  = scores + (size_t)b * SEQL;
    float*       al = alpha  + (size_t)b * SEQL;

    float m = -3.402823466e38f;
    for (int i = tid; i < SEQL; i += 256) m = fmaxf(m, s[i]);
#pragma unroll
    for (int off = 16; off >= 1; off >>= 1) m = fmaxf(m, __shfl_xor(m, off, 32));
    if ((tid & 31) == 0) red[tid >> 5] = m;
    __syncthreads();
    if (tid == 0) {
        float t = red[0];
        for (int j = 1; j < 8; ++j) t = fmaxf(t, red[j]);
        bcast[0] = t;
    }
    __syncthreads();
    float bm = bcast[0];

    float sum = 0.f;
    for (int i = tid; i < SEQL; i += 256) {
        float e = __expf(s[i] - bm);
        al[i] = e;
        sum += e;
    }
#pragma unroll
    for (int off = 16; off >= 1; off >>= 1) sum += __shfl_xor(sum, off, 32);
    __syncthreads();
    if ((tid & 31) == 0) red[tid >> 5] = sum;
    __syncthreads();
    if (tid == 0) {
        float t = 0.f;
        for (int j = 0; j < 8; ++j) t += red[j];
        bcast[1] = 1.0f / t;
    }
    __syncthreads();
    float inv = bcast[1];
    for (int i = tid; i < SEQL; i += 256) al[i] *= inv;
}

// ---------------------------------------------------------------------------
// Kernel 5: weighted partials: 8-way split over L (deterministic, no atomics)
// WG (ls,b): 256 threads x float4 cover all 1024 E columns.
// ---------------------------------------------------------------------------
__global__ __launch_bounds__(256) void weighted_partial_kernel(
    const float* __restrict__ enc, const float* __restrict__ alpha,
    float* __restrict__ partials)
{
    int ls = blockIdx.x, b = blockIdx.y, tid = threadIdx.x;
    const float* base = enc + ((size_t)b * SEQL + ls * 256) * EDIM + tid * 4;
    const float* al   = alpha + (size_t)b * SEQL + ls * 256;
    float ax = 0.f, ay = 0.f, az = 0.f, aw = 0.f;
    for (int l = 0; l < 256; ++l) {
        float a  = al[l];
        float4 v = *(const float4*)(base + (size_t)l * EDIM);
        ax += a * v.x; ay += a * v.y; az += a * v.z; aw += a * v.w;
    }
    float4 o; o.x = ax; o.y = ay; o.z = az; o.w = aw;
    *(float4*)(partials + ((size_t)(ls * BATCH + b)) * EDIM + tid * 4) = o;
}

__global__ __launch_bounds__(256) void weighted_reduce_kernel(
    const float* __restrict__ partials, float* __restrict__ weighted)
{
    int b = blockIdx.x, tid = threadIdx.x;
    float ax = 0.f, ay = 0.f, az = 0.f, aw = 0.f;
#pragma unroll
    for (int ls = 0; ls < 8; ++ls) {
        float4 v = *(const float4*)(partials + ((size_t)(ls * BATCH + b)) * EDIM + tid * 4);
        ax += v.x; ay += v.y; az += v.z; aw += v.w;
    }
    float4 o; o.x = ax; o.y = ay; o.z = az; o.w = aw;
    *(float4*)(weighted + (size_t)b * EDIM + tid * 4) = o;
}

// ---------------------------------------------------------------------------
extern "C" void kernel_launch(void* const* d_in, const int* in_sizes, int n_in,
                              void* d_out, int out_size, void* d_ws, size_t ws_size,
                              hipStream_t stream)
{
    const float* enc  = (const float*)d_in[0];   // [64,2048,1024]
    const float* dh   = (const float*)d_in[1];   // [64,1024]
    const float* We_w = (const float*)d_in[2];   // [1024,512]
    const float* We_b = (const float*)d_in[3];   // [512]
    const float* Wd_w = (const float*)d_in[4];   // [1024,512]
    const float* Wd_b = (const float*)d_in[5];   // [512]
    const float* Wf_w = (const float*)d_in[6];   // [512]
    const float* Wf_b = (const float*)d_in[7];   // [1]

    char* ws = (char*)d_ws;
    bf16_t* Wt      = (bf16_t*)ws;                                   // 1 MiB
    float*  decc    = (float*)(ws + 1048576);                        // 128 KiB
    float*  scores  = (float*)(ws + 1048576 + 131072);               // 512 KiB
    float*  partials= (float*)(ws + 1048576 + 131072 + 524288);     // 2 MiB

    float* weighted = (float*)d_out;                 // [64,1024]
    float* alpha    = weighted + BATCH * EDIM;       // [64,2048]

    we_transpose_kernel<<<dim3(EDIM), dim3(ADIM), 0, stream>>>(We_w, Wt);
    dec_proj_kernel<<<dim3(BATCH), dim3(ADIM), 0, stream>>>(dh, Wd_w, Wd_b, We_b, decc);
    attn_scores_kernel<<<dim3(SEQL / 32, BATCH), dim3(256), 0, stream>>>(
        enc, Wt, decc, Wf_w, Wf_b, scores);
    softmax_kernel<<<dim3(BATCH), dim3(256), 0, stream>>>(scores, alpha);
    weighted_partial_kernel<<<dim3(8, BATCH), dim3(256), 0, stream>>>(enc, alpha, partials);
    weighted_reduce_kernel<<<dim3(BATCH), dim3(256), 0, stream>>>(partials, weighted);
}
